// GryphonHRMLayer_35381940584735
// MI455X (gfx1250) — compile-verified, matrix operand order checked by reading, not run on hardware
//
#include <hip/hip_runtime.h>
#include <hip/hip_bf16.h>

// ---------------- problem constants ----------------
constexpr int BB  = 4;
constexpr int SS  = 4096;
constexpr int GG  = 64;
constexpr int DD  = 1024;
constexpr int HH  = 16;
constexpr int BLKC= 64;
constexpr int PP  = 256;
constexpr int FF  = 4096;
constexpr int HDC = 64;
constexpr int NBSC= 64;
constexpr int TTC = GG + SS;          // 4160
constexpr int CPL = 256;
constexpr int NCC = 64;               // scan chunks
constexpr int CHC = 64;               // chunk length
constexpr float EPSI = 1e-5f;
constexpr float NEGV = -1.0e9f;

#define USE_ASYNC_LDS 1

typedef __attribute__((ext_vector_type(16))) __bf16 v16bf;
typedef __attribute__((ext_vector_type(8)))  float  v8f;

__device__ __forceinline__ float gelu_f(float x) {
    float x3 = x * x * x;
    return 0.5f * x * (1.0f + tanhf(0.7978845608028654f * (x + 0.044715f * x3)));
}

// load a 16x32 bf16 WMMA fragment from LDS; row-major tile with stride `ld` elements.
// lane<16: rows 0..15, K 0..7 & 16..23 ; lane>=16: K 8..15 & 24..31 (per ISA layout).
__device__ __forceinline__ v16bf load_frag(const __bf16* base /* row ptr + kk + kb8 */) {
    const uint4* p = (const uint4*)base;
    v16bf f;
    ((uint4*)&f)[0] = p[0];
    ((uint4*)&f)[1] = p[2];
    return f;
}

// ---------------- generic bf16 WMMA GEMM ----------------
// C[M,N] = act( A[M,K] @ B[K,N] + bias ),  A,B bf16 row-major, f32 accum.
// Tile 128x128, K-step 64. 256 threads = 8 waves (2x4), each wave 64x32 (4x2 wmma tiles).
#define GTM 128
#define GTN 128
#define GTK 64
#define LDT 80   // padded LDS stride (elements); 160B = multiple of 16B

__global__ void __launch_bounds__(256)
gemm_bf16_kernel(const __bf16* __restrict__ A, const __bf16* __restrict__ Bm,
                 const float* __restrict__ bias,
                 float* __restrict__ Cf, __bf16* __restrict__ Cb,
                 int M, int N, int K, int flags)
{
    __shared__ __bf16 sA[GTM * LDT];
    __shared__ __bf16 sB[GTN * LDT];   // stored transposed: [n][k]

    const int tid  = threadIdx.x;
    const int lane = tid & 31;
    const int wave = tid >> 5;
    const int wm   = wave & 1;    // 0..1 -> 64 rows each
    const int wn   = wave >> 1;   // 0..3 -> 32 cols each
    const int m0   = blockIdx.y * GTM;
    const int n0   = blockIdx.x * GTN;

    v8f acc[4][2];
    for (int mi = 0; mi < 4; ++mi)
        for (int ni = 0; ni < 2; ++ni)
            for (int e = 0; e < 8; ++e) acc[mi][ni][e] = 0.0f;

    for (int k0 = 0; k0 < K; k0 += GTK) {
        // ---- A tile: 128x64 bf16 global -> LDS ----
#if USE_ASYNC_LDS
        // CDNA5 async DMA to LDS (ASYNCcnt-tracked), bypasses VGPR round trip.
#pragma unroll
        for (int it = 0; it < 4; ++it) {
            int slot = tid + 256 * it;        // 0..1023
            int row  = slot >> 3;             // 0..127
            int g    = slot & 7;              // 8 bf16 per group
            const __bf16* gptr = A + (size_t)(m0 + row) * K + k0 + g * 8;
            unsigned ldsaddr = (unsigned)(size_t)(sA + row * LDT + g * 8);
            asm volatile("global_load_async_to_lds_b128 %0, %1, off"
                         :: "v"(ldsaddr), "v"(gptr) : "memory");
        }
#else
#pragma unroll
        for (int it = 0; it < 4; ++it) {
            int slot = tid + 256 * it;
            int row  = slot >> 3;
            int g    = slot & 7;
            const uint4* src = (const uint4*)(A + (size_t)(m0 + row) * K + k0 + g * 8);
            *(uint4*)(sA + row * LDT + g * 8) = *src;
        }
#endif
        // ---- B tile: 64x128 bf16 global -> LDS transposed [n][k] ----
#pragma unroll
        for (int it = 0; it < 4; ++it) {
            int slot = tid + 256 * it;
            int row  = slot >> 4;             // k 0..63
            int g    = slot & 15;             // 8 cols per group
            uint4 tdat = *(const uint4*)(Bm + (size_t)(k0 + row) * N + n0 + g * 8);
            const __bf16* tb = (const __bf16*)&tdat;
#pragma unroll
            for (int e = 0; e < 8; ++e) sB[(g * 8 + e) * LDT + row] = tb[e];
        }
        if (k0 + GTK < K)
            __builtin_prefetch(Bm + (size_t)(k0 + GTK + (tid & 63)) * N + n0, 0, 1);
#if USE_ASYNC_LDS
        asm volatile("s_wait_asynccnt 0x0" ::: "memory");
#endif
        __syncthreads();

        const int lo16 = lane & 15;
        const int kb8  = (lane >> 4) << 3;   // 0 or 8 elements
#pragma unroll
        for (int kk = 0; kk < GTK; kk += 32) {
            v16bf af[4];
#pragma unroll
            for (int mi = 0; mi < 4; ++mi)
                af[mi] = load_frag(sA + (wm * 64 + mi * 16 + lo16) * LDT + kk + kb8);
            v16bf bfr[2];
#pragma unroll
            for (int ni = 0; ni < 2; ++ni)
                bfr[ni] = load_frag(sB + (wn * 32 + ni * 16 + lo16) * LDT + kk + kb8);
#pragma unroll
            for (int mi = 0; mi < 4; ++mi)
#pragma unroll
                for (int ni = 0; ni < 2; ++ni)
                    acc[mi][ni] = __builtin_amdgcn_wmma_f32_16x16x32_bf16(
                        false, af[mi], false, bfr[ni], (short)0, acc[mi][ni], false, false);
        }
        __syncthreads();
    }

    // ---- epilogue: C/D layout — VGPR e : M = e + (lane>=16 ? 8 : 0), N = lane&15 ----
    const int lanen = lane & 15;
    const int mo    = (lane >> 4) * 8;
    for (int mi = 0; mi < 4; ++mi) {
        for (int ni = 0; ni < 2; ++ni) {
            int c = n0 + wn * 32 + ni * 16 + lanen;
            float bv = bias ? bias[c] : 0.0f;
#pragma unroll
            for (int e = 0; e < 8; ++e) {
                int r = m0 + wm * 64 + mi * 16 + mo + e;
                float v = acc[mi][ni][e] + bv;
                if (flags & 1) v = gelu_f(v);
                if (Cf) Cf[(size_t)r * N + c] = v;
                if (Cb) Cb[(size_t)r * N + c] = (__bf16)v;
            }
        }
    }
}

// ---------------- weight f32 -> bf16 ----------------
__global__ void wcvt_kernel(const float* __restrict__ src, __bf16* __restrict__ dst, int n) {
    int i = blockIdx.x * 256 + threadIdx.x;
    if (i < n) dst[i] = (__bf16)src[i];
}

// global tokens -> bf16 rows of x
__global__ void gcvt_kernel(const float* __restrict__ gt, __bf16* __restrict__ xb) {
    int i = blockIdx.x * 256 + threadIdx.x;
    if (i >= BB * GG * DD) return;
    int b = i / (GG * DD);
    int rem = i % (GG * DD);
    xb[(size_t)(b * TTC) * DD + rem] = (__bf16)gt[i];
}

// ---------------- RMSNorm (layer 1): f32 + bf16 + x rows ----------------
__global__ void __launch_bounds__(256)
prep_norm_kernel(const float* __restrict__ hs, const float* __restrict__ w,
                 float* __restrict__ normed, __bf16* __restrict__ nrmb, __bf16* __restrict__ xb)
{
    int row = blockIdx.x;             // 0..BB*SS-1
    int tid = threadIdx.x;
    int b = row / SS, s = row % SS;
    const float* x = hs + (size_t)row * DD;
    float vals[4]; float ss = 0.0f;
    for (int i = 0; i < 4; ++i) { float v = x[tid + 256 * i]; vals[i] = v; ss += v * v; }
    __shared__ float red[256];
    red[tid] = ss; __syncthreads();
    for (int s2 = 128; s2 > 0; s2 >>= 1) { if (tid < s2) red[tid] += red[tid + s2]; __syncthreads(); }
    float scale = rsqrtf(red[0] * (1.0f / DD) + EPSI);
    size_t xrow = (size_t)(b * TTC + GG + s) * DD;
    for (int i = 0; i < 4; ++i) {
        int d = tid + 256 * i;
        float y = vals[i] * scale * w[d];
        normed[(size_t)row * DD + d] = y;
        nrmb[(size_t)row * DD + d]   = (__bf16)y;
        xb[xrow + d]                 = (__bf16)y;
    }
}

// ---------------- RMSNorm (layer 2): bf16 only ----------------
__global__ void __launch_bounds__(256)
norm2_kernel(const float* __restrict__ hin, const float* __restrict__ w, __bf16* __restrict__ n2b)
{
    int row = blockIdx.x;
    int tid = threadIdx.x;
    const float* x = hin + (size_t)row * DD;
    float vals[4]; float ss = 0.0f;
    for (int i = 0; i < 4; ++i) { float v = x[tid + 256 * i]; vals[i] = v; ss += v * v; }
    __shared__ float red[256];
    red[tid] = ss; __syncthreads();
    for (int s2 = 128; s2 > 0; s2 >>= 1) { if (tid < s2) red[tid] += red[tid + s2]; __syncthreads(); }
    float scale = rsqrtf(red[0] * (1.0f / DD) + EPSI);
    for (int i = 0; i < 4; ++i) {
        int d = tid + 256 * i;
        n2b[(size_t)row * DD + d] = (__bf16)(vals[i] * scale * w[d]);
    }
}

// ---------------- cumulative positions per batch ----------------
__global__ void pos_kernel(const float* __restrict__ mask, int* __restrict__ pos) {
    int b = blockIdx.x;
    int tid = threadIdx.x;          // 256 threads, 16 elements each
    __shared__ int part[256];
    int base = tid * 16;
    int loc[16]; int cnt = 0;
    for (int i = 0; i < 16; ++i) {
        int m = mask[b * SS + base + i] > 0.5f ? 1 : 0;
        loc[i] = m; cnt += m;
    }
    part[tid] = cnt; __syncthreads();
    for (int off = 1; off < 256; off <<= 1) {
        int v = (tid >= off) ? part[tid - off] : 0;
        __syncthreads();
        part[tid] += v;
        __syncthreads();
    }
    int run = part[tid] - cnt;      // exclusive prefix
    for (int i = 0; i < 16; ++i) { run += loc[i]; pos[b * SS + base + i] = run - 1; }
}

// ---------------- RoPE on q and k (in place, f32) ----------------
__global__ void rope_kernel(float* __restrict__ q, float* __restrict__ k, const int* __restrict__ pos) {
    int idx = blockIdx.x * 256 + threadIdx.x;
    if (idx >= BB * TTC * HH * 32) return;
    int j  = idx & 31;
    int h  = (idx >> 5) & 15;
    int bt = idx >> 9;
    int t  = bt % TTC;
    int b  = bt / TTC;
    int rp = 0;
    if (t >= GG) { int p0 = pos[b * SS + t - GG]; rp = p0 > 0 ? p0 : 0; }
    float inv = expf(-(float)j * (9.210340371976184f / 32.0f));   // 10000^(-j/32)
    float ang = (float)rp * inv;
    float cs = cosf(ang), sn = sinf(ang);
    size_t base = (size_t)(b * TTC + t) * DD + h * HDC + j;
    float x1 = q[base], x2 = q[base + 32];
    q[base] = x1 * cs - x2 * sn; q[base + 32] = x1 * sn + x2 * cs;
    x1 = k[base]; x2 = k[base + 32];
    k[base] = x1 * cs - x2 * sn; k[base + 32] = x1 * sn + x2 * cs;
}

// ---------------- global-token attention: one block per (b,h,qg) ----------------
__global__ void __launch_bounds__(256)
gattn_kernel(const float* __restrict__ q, const float* __restrict__ k, const float* __restrict__ v,
             const float* __restrict__ mask, __bf16* __restrict__ attnb)
{
    __shared__ float s_q[HDC];
    __shared__ float s_sc[TTC];
    __shared__ float s_red[256];
    __shared__ float s_stat[2];
    int bid = blockIdx.x;
    int qg = bid % GG;
    int h  = (bid / GG) % HH;
    int b  = bid / (GG * HH);
    int tid = threadIdx.x;

    size_t qoff = (size_t)(b * TTC + qg) * DD + h * HDC;
    if (tid < HDC) s_q[tid] = q[qoff + tid];
    __syncthreads();

    float lmax = -3.0e38f;
    for (int t = tid; t < TTC; t += 256) {
        bool ok = (t < GG) || (mask[b * SS + (t - GG)] > 0.5f);
        float sc = NEGV;
        if (ok) {
            const float* kr = k + (size_t)(b * TTC + t) * DD + h * HDC;
            float d0 = 0.0f;
            for (int j = 0; j < HDC; ++j) d0 += s_q[j] * kr[j];
            sc = d0 * 0.125f;
        }
        s_sc[t] = sc;
        lmax = fmaxf(lmax, sc);
    }
    s_red[tid] = lmax; __syncthreads();
    for (int s2 = 128; s2 > 0; s2 >>= 1) { if (tid < s2) s_red[tid] = fmaxf(s_red[tid], s_red[tid + s2]); __syncthreads(); }
    if (tid == 0) s_stat[0] = s_red[0];
    __syncthreads();
    float mx = s_stat[0];

    float lsum = 0.0f;
    for (int t = tid; t < TTC; t += 256) { float e = expf(s_sc[t] - mx); s_sc[t] = e; lsum += e; }
    s_red[tid] = lsum; __syncthreads();
    for (int s2 = 128; s2 > 0; s2 >>= 1) { if (tid < s2) s_red[tid] += s_red[tid + s2]; __syncthreads(); }
    if (tid == 0) s_stat[1] = s_red[0];
    __syncthreads();
    float inv = 1.0f / s_stat[1];

    int j = tid & 63, quad = tid >> 6;
    float part = 0.0f;
    for (int t = quad; t < TTC; t += 4)
        part += s_sc[t] * v[(size_t)(b * TTC + t) * DD + h * HDC + j];
    s_red[tid] = part; __syncthreads();
    if (tid < 64) {
        float o = (s_red[tid] + s_red[tid + 64] + s_red[tid + 128] + s_red[tid + 192]) * inv;
        attnb[(size_t)(b * TTC + qg) * DD + h * HDC + tid] = (__bf16)o;
    }
}

// ---------------- local block attention (WMMA): one block per (b,nb,h) ----------------
// S = Q[64,64] Kt[64,256] via wmma; softmax rows; O = P[64,256] V[256,64] via wmma.
constexpr int LAT_SQ  = 64 * 72;      // bf16
constexpr int LAT_SK  = 256 * 72;     // bf16
constexpr int LAT_SVT = 64 * 264;     // bf16 (V transposed: [j][key])
constexpr int LAT_SSC = 64 * 256;     // f32 scores
constexpr int LAT_SP  = 64 * 264;     // bf16 probs
constexpr size_t LAT_SMEM =
    (size_t)(LAT_SQ + LAT_SK + LAT_SVT + LAT_SP) * 2 + (size_t)LAT_SSC * 4 +
    (size_t)(256 + 256 + 64) * 4;

__global__ void __launch_bounds__(256)
lattn_kernel(const float* __restrict__ q, const float* __restrict__ k, const float* __restrict__ v,
             const float* __restrict__ mask, const int* __restrict__ pos, __bf16* __restrict__ attnb)
{
    extern __shared__ char smem[];
    __bf16* s_q  = (__bf16*)smem;                         // [64][72]
    __bf16* s_k  = s_q + LAT_SQ;                          // [256][72]
    __bf16* s_vT = s_k + LAT_SK;                          // [64][264]
    float*  s_sc = (float*)(s_vT + LAT_SVT);              // [64][256]
    __bf16* s_p  = (__bf16*)(s_sc + LAT_SSC);             // [64][264]
    int* s_krow = (int*)(s_p + LAT_SP);                   // 256
    int* s_kp   = s_krow + 256;                           // 256
    int* s_qpos = s_kp + 256;                             // 64

    int bid = blockIdx.x;
    int h  = bid % HH;
    int nb = (bid / HH) % NBSC;
    int b  = bid / (HH * NBSC);
    int tid  = threadIdx.x;
    int lane = tid & 31;
    int wave = tid >> 5;

    // ---- key map + q tile ----
    {
        int kidx = tid;
        int krow, kp;
        if (kidx < GG) { krow = b * TTC + kidx; kp = -1; }
        else {
            int jj = kidx - GG;
            int blk = nb + (jj >> 6) - 1;
            int sk  = blk * BLKC + (jj & 63);
            if (blk >= 0 && blk < NBSC && mask[b * SS + sk] > 0.5f) {
                krow = b * TTC + GG + sk; kp = pos[b * SS + sk];
            } else { krow = -1; kp = 0; }
        }
        s_krow[kidx] = krow; s_kp[kidx] = kp;
    }
    if (tid < 64) s_qpos[tid] = pos[b * SS + nb * BLKC + tid];
    for (int e = tid; e < 64 * 64; e += 256) {
        int r = e >> 6, j = e & 63;
        s_q[r * 72 + j] = (__bf16)q[(size_t)(b * TTC + GG + nb * BLKC + r) * DD + h * HDC + j];
    }
    __syncthreads();

    // ---- K and V tiles (invalid keys zero-filled) ----
    for (int e = tid; e < 256 * 64; e += 256) {
        int r = e >> 6, j = e & 63;     // r = key idx, j = head dim
        int krow = s_krow[r];
        float kv = 0.0f, vv = 0.0f;
        if (krow >= 0) {
            size_t o = (size_t)krow * DD + h * HDC;
            kv = k[o + j]; vv = v[o + j];
        }
        s_k[r * 72 + j]   = (__bf16)kv;
        s_vT[j * 264 + r] = (__bf16)vv;
    }
    __syncthreads();

    const int lo16 = lane & 15;
    const int kb8  = (lane >> 4) << 3;
    const int mo   = (lane >> 4) * 8;

    // ---- scores: each wave does 16 rows x 128 cols (8 n-tiles) ----
    {
        int wm  = wave >> 1;   // 0..3
        int wn2 = wave & 1;    // 0..1
        v8f acc[8];
        for (int nt = 0; nt < 8; ++nt)
            for (int e = 0; e < 8; ++e) acc[nt][e] = 0.0f;
#pragma unroll
        for (int kk = 0; kk < 64; kk += 32) {
            v16bf af = load_frag(s_q + (wm * 16 + lo16) * 72 + kk + kb8);
#pragma unroll
            for (int nt = 0; nt < 8; ++nt) {
                v16bf bf = load_frag(s_k + (wn2 * 128 + nt * 16 + lo16) * 72 + kk + kb8);
                acc[nt] = __builtin_amdgcn_wmma_f32_16x16x32_bf16(
                    false, af, false, bf, (short)0, acc[nt], false, false);
            }
        }
        // scale + mask, write to LDS
#pragma unroll
        for (int nt = 0; nt < 8; ++nt) {
            int c = wn2 * 128 + nt * 16 + lo16;
            bool kvalid = s_krow[c] >= 0;
            int  kp = s_kp[c];
#pragma unroll
            for (int e = 0; e < 8; ++e) {
                int r = wm * 16 + mo + e;
                bool allow = kvalid && (kp <= s_qpos[r]);
                s_sc[r * 256 + c] = allow ? acc[nt][e] * 0.125f : NEGV;
            }
        }
    }
    __syncthreads();

    // ---- softmax per row, emit bf16 probs ----
    if (tid < 64) {
        float* row = s_sc + tid * 256;
        float mx = -3.0e38f;
        for (int i = 0; i < 256; ++i) mx = fmaxf(mx, row[i]);
        float sum = 0.0f;
        float ex[0 ? 0 : 1];  // (placeholder to keep loop simple)
        (void)ex;
        for (int i = 0; i < 256; ++i) { float e2 = expf(row[i] - mx); row[i] = e2; sum += e2; }
        float inv = 1.0f / sum;
        __bf16* prow = s_p + tid * 264;
        for (int i = 0; i < 256; ++i) prow[i] = (__bf16)(row[i] * inv);
    }
    __syncthreads();

    // ---- O = P @ V : 16 tiles, 2 per wave ----
    {
        int mt = wave >> 1;    // 0..3
        int np = wave & 1;     // 0..1 -> n-tiles np*2, np*2+1
        v8f oacc[2];
        for (int t = 0; t < 2; ++t)
            for (int e = 0; e < 8; ++e) oacc[t][e] = 0.0f;
#pragma unroll
        for (int kk = 0; kk < 256; kk += 32) {
            v16bf af = load_frag(s_p + (mt * 16 + lo16) * 264 + kk + kb8);
#pragma unroll
            for (int t = 0; t < 2; ++t) {
                v16bf bf = load_frag(s_vT + ((np * 2 + t) * 16 + lo16) * 264 + kk + kb8);
                oacc[t] = __builtin_amdgcn_wmma_f32_16x16x32_bf16(
                    false, af, false, bf, (short)0, oacc[t], false, false);
            }
        }
#pragma unroll
        for (int t = 0; t < 2; ++t) {
            int c = (np * 2 + t) * 16 + lo16;
#pragma unroll
            for (int e = 0; e < 8; ++e) {
                int r = mt * 16 + mo + e;
                attnb[(size_t)(b * TTC + GG + nb * BLKC + r) * DD + h * HDC + c] = (__bf16)oacc[t][e];
            }
        }
    }
}

// ---------------- copy new_global rows from attn output ----------------
__global__ void newglobal_kernel(const float* __restrict__ attno, float* __restrict__ outg) {
    int i = blockIdx.x * 256 + threadIdx.x;
    if (i >= BB * GG * DD) return;
    int b = i / (GG * DD);
    int rem = i % (GG * DD);
    outg[i] = attno[(size_t)(b * TTC) * DD + rem];
}

// ---------------- S5 discretization: BdT[ D x 2P ], Ccomb[ 2P x D ], Lam_d, Lam_d^64 ----------------
__global__ void s5prep_kernel(const float* __restrict__ Lre, const float* __restrict__ Lim,
                              const float* __restrict__ Bre, const float* __restrict__ Bim,
                              const float* __restrict__ Cre, const float* __restrict__ Cim,
                              const float* __restrict__ logdt,
                              __bf16* __restrict__ bdT, __bf16* __restrict__ ccomb,
                              float* __restrict__ lam)
{
    int idx = blockIdx.x * 256 + threadIdx.x;
    if (idx >= PP * DD) return;
    int p = idx & 255;
    int d = idx >> 8;
    float lr = Lre[p], li = Lim[p];
    float dt = expf(logdt[p]);
    float mag = expf(lr * dt);
    float ar = mag * cosf(li * dt), ai = mag * sinf(li * dt);
    float den = lr * lr + li * li;
    float nr = ar - 1.0f, ni = ai;
    float cr  = (nr * lr + ni * li) / den;
    float cim = (ni * lr - nr * li) / den;
    float br = Bre[(size_t)p * DD + d], bi = Bim[(size_t)p * DD + d];
    bdT[(size_t)d * 512 + p]       = (__bf16)(cr * br - cim * bi);
    bdT[(size_t)d * 512 + 256 + p] = (__bf16)(cr * bi + cim * br);
    ccomb[(size_t)p * DD + d]         = (__bf16)( 2.0f * Cre[(size_t)d * PP + p]);
    ccomb[(size_t)(256 + p) * DD + d] = (__bf16)(-2.0f * Cim[(size_t)d * PP + p]);
    if (d == 0) {
        lam[p * 4] = ar; lam[p * 4 + 1] = ai;
        float sr = ar, si = ai;
        for (int i = 0; i < 6; ++i) { float tr = sr * sr - si * si, ti = 2.0f * sr * si; sr = tr; si = ti; }
        lam[p * 4 + 2] = sr; lam[p * 4 + 3] = si;   // a^64
    }
}

// ---------------- S5 chunked scan: phase 1 (chunk local, zero init) ----------------
__global__ void scan_chunk_kernel(const float* __restrict__ Bu, const float* __restrict__ lam,
                                  float* __restrict__ cF)
{
    int idx = blockIdx.x * 256 + threadIdx.x;
    if (idx >= BB * PP * NCC) return;
    int p = idx & 255, c = (idx >> 8) & 63, b = idx >> 14;
    float ar = lam[p * 4], ai = lam[p * 4 + 1];
    float xr = 0.0f, xi = 0.0f;
    for (int i = 0; i < CHC; ++i) {
        size_t row = (size_t)(b * SS + c * CHC + i) * 512;
        float br = Bu[row + p], bi = Bu[row + 256 + p];
        float ny = ar * xi + ai * xr + bi;
        xr = ar * xr - ai * xi + br;
        xi = ny;
    }
    size_t o = ((size_t)((b * PP + p) * NCC + c)) * 2;
    cF[o] = xr; cF[o + 1] = xi;
}

// ---------------- S5 scan: phase 2 (across chunks with a^64) ----------------
__global__ void scan_prefix_kernel(const float* __restrict__ cF, const float* __restrict__ lam,
                                   const float* __restrict__ s5st, float* __restrict__ cS)
{
    int idx = blockIdx.x * 256 + threadIdx.x;
    if (idx >= BB * PP) return;
    int p = idx & 255, b = idx >> 8;
    float a64r = lam[p * 4 + 2], a64i = lam[p * 4 + 3];
    float xr = s5st[(size_t)(b * PP + p) * 2];
    float xi = s5st[(size_t)(b * PP + p) * 2 + 1];
    for (int c = 0; c < NCC; ++c) {
        size_t o = ((size_t)((b * PP + p) * NCC + c)) * 2;
        cS[o] = xr; cS[o + 1] = xi;                // state entering chunk c
        float fr = cF[o], fi = cF[o + 1];
        float ny = a64r * xi + a64i * xr + fi;
        xr = a64r * xr - a64i * xi + fr;
        xi = ny;
    }
}

// ---------------- S5 scan: phase 3 (replay with true initial state, emit xs bf16) ----------------
__global__ void scan_final_kernel(const float* __restrict__ Bu, const float* __restrict__ lam,
                                  const float* __restrict__ cS, __bf16* __restrict__ xsb,
                                  float* __restrict__ outState)
{
    int idx = blockIdx.x * 256 + threadIdx.x;
    if (idx >= BB * PP * NCC) return;
    int p = idx & 255, c = (idx >> 8) & 63, b = idx >> 14;
    float ar = lam[p * 4], ai = lam[p * 4 + 1];
    size_t so = ((size_t)((b * PP + p) * NCC + c)) * 2;
    float xr = cS[so], xi = cS[so + 1];
    for (int i = 0; i < CHC; ++i) {
        size_t row = (size_t)(b * SS + c * CHC + i) * 512;
        float br = Bu[row + p], bi = Bu[row + 256 + p];
        float ny = ar * xi + ai * xr + bi;
        xr = ar * xr - ai * xi + br;
        xi = ny;
        xsb[row + p]       = (__bf16)xr;
        xsb[row + 256 + p] = (__bf16)xi;
    }
    if (c == NCC - 1) {
        outState[(size_t)(b * PP + p) * 2]     = xr;
        outState[(size_t)(b * PP + p) * 2 + 1] = xi;
    }
}

// ---------------- gate head: g1 @ Wg2 + bg2 -> sigmoid ----------------
__global__ void gate2_kernel(const __bf16* __restrict__ g1, const float* __restrict__ Wg2,
                             const float* __restrict__ bg2, float* __restrict__ gates)
{
    int r = blockIdx.x * 256 + threadIdx.x;
    if (r >= BB * SS) return;
    const __bf16* row = g1 + (size_t)r * CPL;
    float a0 = 0.0f, a1 = 0.0f;
    for (int i = 0; i < CPL; ++i) {
        float g = (float)row[i];
        a0 += g * Wg2[i * 2];
        a1 += g * Wg2[i * 2 + 1];
    }
    a0 += bg2[0]; a1 += bg2[1];
    gates[(size_t)r * 2]     = 1.0f / (1.0f + expf(-a0));
    gates[(size_t)r * 2 + 1] = 1.0f / (1.0f + expf(-a1));
}

// ---------------- fuse: h = hidden + g0*bb + g1*(s5gemm + normed*D_skip) ----------------
__global__ void fuse_kernel(const float* __restrict__ hs, const float* __restrict__ attno,
                            const float* __restrict__ s5g, const float* __restrict__ normed,
                            const float* __restrict__ Dskip, const float* __restrict__ gates,
                            float* __restrict__ hout)
{
    size_t i = (size_t)blockIdx.x * 256 + threadIdx.x;
    if (i >= (size_t)BB * SS * DD) return;
    int row = (int)(i / DD);
    int d   = (int)(i % DD);
    int b = row / SS, s = row % SS;
    float bb  = attno[(size_t)(b * TTC + GG + s) * DD + d];
    float s5o = s5g[i] + normed[i] * Dskip[d];
    float g0 = gates[(size_t)row * 2], g1 = gates[(size_t)row * 2 + 1];
    hout[i] = hs[i] + g0 * bb + g1 * s5o;
}

// ---------------- final residual ----------------
__global__ void final_kernel(const float* __restrict__ h, const float* __restrict__ f2,
                             float* __restrict__ out)
{
    size_t i = (size_t)blockIdx.x * 256 + threadIdx.x;
    if (i >= (size_t)BB * SS * DD) return;
    out[i] = h[i] + f2[i];
}

// =====================================================================
extern "C" void kernel_launch(void* const* d_in, const int* in_sizes, int n_in,
                              void* d_out, int out_size, void* d_ws, size_t ws_size,
                              hipStream_t stream)
{
    (void)in_sizes; (void)n_in; (void)out_size; (void)ws_size;
    const float* hidden = (const float*)d_in[0];
    const float* gtok   = (const float*)d_in[1];
    const float* amask  = (const float*)d_in[2];
    const float* s5st   = (const float*)d_in[3];
    const float* ln1w   = (const float*)d_in[4];
    const float* ln2w   = (const float*)d_in[5];
    const float* Wq     = (const float*)d_in[6];
    const float* Wk     = (const float*)d_in[7];
    const float* Wv     = (const float*)d_in[8];
    const float* Wo     = (const float*)d_in[9];
    const float* Lre    = (const float*)d_in[10];
    const float* Lim    = (const float*)d_in[11];
    const float* Bre    = (const float*)d_in[12];
    const float* Bim    = (const float*)d_in[13];
    const float* Cre    = (const float*)d_in[14];
    const float* Cim    = (const float*)d_in[15];
    const float* Dskip  = (const float*)d_in[16];
    const float* logdt  = (const float*)d_in[17];
    const float* Wg1    = (const float*)d_in[18];
    const float* bg1    = (const float*)d_in[19];
    const float* Wg2    = (const float*)d_in[20];
    const float* bg2    = (const float*)d_in[21];
    const float* W1     = (const float*)d_in[22];
    const float* W2     = (const float*)d_in[23];

    float* out = (float*)d_out;
    const int BT = BB * TTC;   // 16640
    const int BS = BB * SS;    // 16384

    char* ws = (char*)d_ws;
    size_t off = 0;
    auto alloc = [&](size_t bytes) -> char* {
        size_t o = (off + 255) & ~(size_t)255;
        off = o + bytes;
        return ws + o;
    };

    float*  normed = (float*) alloc((size_t)BS * DD * 4);
    __bf16* nrmb   = (__bf16*)alloc((size_t)BS * DD * 2);
    __bf16* xb     = (__bf16*)alloc((size_t)BT * DD * 2);
    float*  qf     = (float*) alloc((size_t)BT * DD * 4);
    float*  kf     = (float*) alloc((size_t)BT * DD * 4);
    float*  vf     = (float*) alloc((size_t)BT * DD * 4);
    __bf16* attnb  = (__bf16*)alloc((size_t)BT * DD * 2);
    float*  attno  = (float*) alloc((size_t)BT * DD * 4);
    __bf16* wqb    = (__bf16*)alloc((size_t)DD * DD * 2);
    __bf16* wkb    = (__bf16*)alloc((size_t)DD * DD * 2);
    __bf16* wvb    = (__bf16*)alloc((size_t)DD * DD * 2);
    __bf16* wob    = (__bf16*)alloc((size_t)DD * DD * 2);
    __bf16* w1b    = (__bf16*)alloc((size_t)DD * FF * 2);
    __bf16* w2b    = (__bf16*)alloc((size_t)FF * DD * 2);
    __bf16* wg1b   = (__bf16*)alloc((size_t)DD * CPL * 2);
    __bf16* bdT    = (__bf16*)alloc((size_t)DD * 512 * 2);
    __bf16* ccomb  = (__bf16*)alloc((size_t)512 * DD * 2);
    float*  lamd   = (float*) alloc((size_t)PP * 4 * 4);
    int*    posb   = (int*)   alloc((size_t)BB * SS * 4);
    float*  Buf    = (float*) alloc((size_t)BS * 512 * 4);
    __bf16* xsb    = (__bf16*)alloc((size_t)BS * 512 * 2);
    float*  cF     = (float*) alloc((size_t)BB * PP * NCC * 2 * 4);
    float*  cS     = (float*) alloc((size_t)BB * PP * NCC * 2 * 4);
    __bf16* g1b    = (__bf16*)alloc((size_t)BS * CPL * 2);
    float*  gates  = (float*) alloc((size_t)BS * 2 * 4);
    float*  s5g    = (float*) alloc((size_t)BS * DD * 4);
    float*  hbuf   = (float*) alloc((size_t)BS * DD * 4);
    __bf16* n2b    = (__bf16*)alloc((size_t)BS * DD * 2);
    __bf16* h1b    = (__bf16*)alloc((size_t)BS * FF * 2);
    float*  ffn2   = (float*) alloc((size_t)BS * DD * 4);

    const int NT = 256;
#define CDIV(a, b) (((a) + (b) - 1) / (b))

    // weight conversions
    wcvt_kernel<<<CDIV(DD * DD, NT), NT, 0, stream>>>(Wq, wqb, DD * DD);
    wcvt_kernel<<<CDIV(DD * DD, NT), NT, 0, stream>>>(Wk, wkb, DD * DD);
    wcvt_kernel<<<CDIV(DD * DD, NT), NT, 0, stream>>>(Wv, wvb, DD * DD);
    wcvt_kernel<<<CDIV(DD * DD, NT), NT, 0, stream>>>(Wo, wob, DD * DD);
    wcvt_kernel<<<CDIV(DD * FF, NT), NT, 0, stream>>>(W1, w1b, DD * FF);
    wcvt_kernel<<<CDIV(FF * DD, NT), NT, 0, stream>>>(W2, w2b, FF * DD);
    wcvt_kernel<<<CDIV(DD * CPL, NT), NT, 0, stream>>>(Wg1, wg1b, DD * CPL);

    // S5 discretized matrices + Lam_d powers
    s5prep_kernel<<<CDIV(PP * DD, NT), NT, 0, stream>>>(Lre, Lim, Bre, Bim, Cre, Cim, logdt,
                                                        bdT, ccomb, lamd);
    // positions
    pos_kernel<<<BB, NT, 0, stream>>>(amask, posb);

    // RMSNorm 1 (+ bf16 copies); global tokens into x
    prep_norm_kernel<<<BS, NT, 0, stream>>>(hidden, ln1w, normed, nrmb, xb);
    gcvt_kernel<<<CDIV(BB * GG * DD, NT), NT, 0, stream>>>(gtok, xb);

    // QKV projections (WMMA)
    gemm_bf16_kernel<<<dim3(DD / GTN, BT / GTM), NT, 0, stream>>>(xb, wqb, nullptr, qf, nullptr, BT, DD, DD, 0);
    gemm_bf16_kernel<<<dim3(DD / GTN, BT / GTM), NT, 0, stream>>>(xb, wkb, nullptr, kf, nullptr, BT, DD, DD, 0);
    gemm_bf16_kernel<<<dim3(DD / GTN, BT / GTM), NT, 0, stream>>>(xb, wvb, nullptr, vf, nullptr, BT, DD, DD, 0);

    // RoPE
    rope_kernel<<<CDIV(BB * TTC * HH * 32, NT), NT, 0, stream>>>(qf, kf, posb);

    // attention
    gattn_kernel<<<BB * HH * GG, NT, 0, stream>>>(qf, kf, vf, amask, attnb);
    lattn_kernel<<<BB * NBSC * HH, NT, LAT_SMEM, stream>>>(qf, kf, vf, amask, posb, attnb);

    // output projection (WMMA)
    gemm_bf16_kernel<<<dim3(DD / GTN, BT / GTM), NT, 0, stream>>>(attnb, wob, nullptr, attno, nullptr, BT, DD, DD, 0);

    // new_global output
    newglobal_kernel<<<CDIV(BB * GG * DD, NT), NT, 0, stream>>>(attno, out + (size_t)BS * DD);

    // gate MLP: normed @ Wg1 + bg1 -> gelu (WMMA, fused epilogue) -> @ Wg2 -> sigmoid
    gemm_bf16_kernel<<<dim3(CPL / GTN, BS / GTM), NT, 0, stream>>>(nrmb, wg1b, bg1, nullptr, g1b, BS, CPL, DD, 1);
    gate2_kernel<<<CDIV(BS, NT), NT, 0, stream>>>(g1b, Wg2, bg2, gates);

    // S5: Bu = normed @ BdT (re|im packed, WMMA)
    gemm_bf16_kernel<<<dim3(512 / GTN, BS / GTM), NT, 0, stream>>>(nrmb, bdT, nullptr, Buf, nullptr, BS, 512, DD, 0);
    // chunked complex scan
    scan_chunk_kernel<<<CDIV(BB * PP * NCC, NT), NT, 0, stream>>>(Buf, lamd, cF);
    scan_prefix_kernel<<<CDIV(BB * PP, NT), NT, 0, stream>>>(cF, lamd, s5st, cS);
    scan_final_kernel<<<CDIV(BB * PP * NCC, NT), NT, 0, stream>>>(Buf, lamd, cS, xsb,
                                                                  out + (size_t)BS * DD + (size_t)BB * GG * DD);
    // s5_out = 2*Re(xs @ Cc^T) via packed real GEMM (WMMA)
    gemm_bf16_kernel<<<dim3(DD / GTN, BS / GTM), NT, 0, stream>>>(xsb, ccomb, nullptr, s5g, nullptr, BS, DD, 512, 0);

    // fuse + residual, RMSNorm 2
    fuse_kernel<<<CDIV(BS * DD, NT), NT, 0, stream>>>(hidden, attno, s5g, normed, Dskip, gates, hbuf);
    norm2_kernel<<<BS, NT, 0, stream>>>(hbuf, ln2w, n2b);

    // FFN (WMMA, gelu fused into first GEMM)
    gemm_bf16_kernel<<<dim3(FF / GTN, BS / GTM), NT, 0, stream>>>(n2b, w1b, nullptr, nullptr, h1b, BS, FF, DD, 1);
    gemm_bf16_kernel<<<dim3(DD / GTN, BS / GTM), NT, 0, stream>>>(h1b, w2b, nullptr, ffn2, nullptr, BS, DD, FF, 0);

    // out = h + ffn
    final_kernel<<<CDIV(BS * DD, NT), NT, 0, stream>>>(hbuf, ffn2, out);
#undef CDIV
}
